// Encoder_79405355369052
// MI455X (gfx1250) — matrix operand order repaired and checked by reading, not compiled, not run on hardware
//
#include <hip/hip_runtime.h>
#include <hip/hip_bf16.h>

// ---------------- problem constants (match reference) ----------------
#define SEQ     1024
#define EMBED   300
#define KPAD    320          // EMBED padded to multiple of 32 for K-tiling
#define KT      (KPAD / 32)  // 10 K-steps
#define HIDDEN  512
#define GATES   2048         // 4*HIDDEN, PyTorch gate order i,f,g,o
#define PAD_IDX 1
#define NBLK    16           // persistent workgroups for the recurrence
#define RTHREADS 512

typedef __attribute__((ext_vector_type(16))) _Float16 v16h;
typedef __attribute__((ext_vector_type(8)))  float    v8f;

union Frag16 { v16h v; unsigned int w[8]; uint4 q[2]; };

__device__ __forceinline__ float sigm(float x) { return 1.0f / (1.0f + __expf(-x)); }

// ============================================================================
// Kernel 1: pre-swizzle W_ih fp32 [GATES][EMBED] into per-lane WMMA B-fragment
// layout, f16. Fragment (tn, kt): 32 lanes x 16 halves contiguous:
//   Wf[(((tn*KT + kt)*32) + lane)*16 + e] = W_ih[n][k],
//   n = tn*16 + (lane&15), k = kt*32 + 16*(e>>3) + 8*(lane>>4) + (e&7)    (k >= EMBED -> 0)
// The GEMM then loads each B fragment as two b128s per lane.
// ============================================================================
__global__ void k_pack_wih(const float* __restrict__ Wih, _Float16* __restrict__ Wf) {
    int idx = blockIdx.x * blockDim.x + threadIdx.x;   // over KPAD*GATES
    if (idx >= KPAD * GATES) return;
    int e    = idx & 15;
    int lane = (idx >> 4) & 31;
    int u    = idx >> 9;               // tn*KT + kt
    int kt   = u % KT;
    int tn   = u / KT;
    int n = (tn << 4) + (lane & 15);
    int k = (kt << 5) + ((e >> 3) << 4) + ((lane >> 4) << 3) + (e & 7);
    float v = (k < EMBED) ? Wih[n * EMBED + k] : 0.0f;
    Wf[idx] = (_Float16)v;
}

// ============================================================================
// Kernel 2: embedding gather fp32 -> f16 xs[SEQ][KPAD]  (k>=EMBED zeroed)
// ============================================================================
__global__ void k_gather(const int* __restrict__ tok, const float* __restrict__ emb,
                         _Float16* __restrict__ xs) {
    int idx = blockIdx.x * blockDim.x + threadIdx.x;   // over SEQ*KPAD
    if (idx >= SEQ * KPAD) return;
    int t = idx / KPAD, k = idx - t * KPAD;
    float v = (k < EMBED) ? emb[(size_t)tok[t] * EMBED + k] : 0.0f;
    xs[idx] = (_Float16)v;
}

// ============================================================================
// Kernel 3: x_gates[SEQ][GATES] = xs @ W_ih.T + b_ih via V_WMMA_F32_16X16X32_F16.
// One wave handles 1 M-tile x 4 N-tiles (A fragment amortized over 4 WMMAs).
//   A (16x32 f16): lane L: M = m0+(L&15); g=L>>4: VGPR i<4 = K pair (2i+8g),
//                  VGPR 4+i = K pair (16+2i+8g) -> two contiguous b128s.
//   B: prepacked fragments -> two b128s per lane.
//   C/D: VGPR r -> row m0 + r + 8g, col n0+(L&15).
// ============================================================================
__global__ void k_xgates(const _Float16* __restrict__ xs, const _Float16* __restrict__ Wf,
                         const float* __restrict__ b_ih, float* __restrict__ xg) {
    const int lane = threadIdx.x & 31;
    const int wave = threadIdx.x >> 5;
    const int task = blockIdx.x * (blockDim.x >> 5) + wave;  // 64*32 = 2048 tasks
    const int tm = task >> 5;            // 0..63   (SEQ/16)
    const int tg = task & 31;            // group of 4 N-tiles
    const int m0 = tm << 4;
    const int g  = lane >> 4;
    const int ml = lane & 15;

    const unsigned int* xrow = (const unsigned int*)(xs + (size_t)(m0 + ml) * KPAD);
    const uint4* Wf4 = (const uint4*)Wf;

    v8f acc[4] = {{}, {}, {}, {}};
#pragma unroll
    for (int kt = 0; kt < KT; ++kt) {
        Frag16 a;
        const unsigned int* xr = xrow + (kt << 4);
#pragma unroll
        for (int i = 0; i < 4; ++i) {
            a.w[i]     = xr[i + 4 * g];        // K = 32kt + 2i + 8g, +1
            a.w[4 + i] = xr[8 + i + 4 * g];    // K = 32kt + 16 + 2i + 8g, +1
        }
#pragma unroll
        for (int u = 0; u < 4; ++u) {
            const int tn = (tg << 2) + u;
            Frag16 b;
            const uint4* bp = Wf4 + ((size_t)(tn * KT + kt) * 32 + lane) * 2;
            b.q[0] = bp[0];
            b.q[1] = bp[1];
            acc[u] = __builtin_amdgcn_wmma_f32_16x16x32_f16(
                /*neg_a=*/false, a.v, /*neg_b=*/false, b.v,
                /*c_mod=*/(short)0, acc[u], /*reuse_a=*/false, /*reuse_b=*/false);
        }
    }
#pragma unroll
    for (int u = 0; u < 4; ++u) {
        const int n = (tg << 6) + (u << 4) + ml;
        const float bias = b_ih[n];
#pragma unroll
        for (int r = 0; r < 8; ++r)
            xg[(size_t)(m0 + r + 8 * g) * GATES + n] = acc[u][r] + bias;
    }
}

// ============================================================================
// Kernel 4: init double-buffered h and the grid-barrier counter (every call,
// stream-ordered, so kernel_launch stays deterministic across graph replays).
// ============================================================================
__global__ void k_init(const float* __restrict__ h0, float* __restrict__ hbuf,
                       unsigned* __restrict__ bar) {
    int tid = threadIdx.x;
    if (tid < HIDDEN) hbuf[tid] = h0[tid];
    if (tid == 0) *bar = 0u;
}

// ============================================================================
// Kernel 5: sequential LSTM recurrence. 16 persistent WGs x 512 threads.
// WG b owns h/c indices j in [32b, 32b+32) -> 128 rows of W_hh (4 gates x 32).
// Wave w (0..15): gate = w>>2, k-part p = w&3, local row = gate*32+lane.
// Each thread keeps its 128 W_hh weights in 32 float4 VGPRs. Per step:
//  - h (512 f32) pulled memory->LDS with GLOBAL_LOAD_ASYNC_TO_LDS_B32
//    (ASYNCcnt-tracked, no VGPR round trip), s_wait_asynccnt 0,
//  - 128 FMAs/thread against uniform-address float4 LDS reads (broadcast),
//  - 4-way partial reduce via LDS, gate math on 32 state threads,
//  - global_prefetch_b8 of next step's x_gates row,
//  - monotonic release/acquire grid barrier; h double-buffered in scratch.
// ============================================================================
__global__ void k_recur(const int*   __restrict__ tok,
                        const float* __restrict__ h0,
                        const float* __restrict__ c0,
                        const float* __restrict__ xg,
                        const float* __restrict__ Whh,
                        const float* __restrict__ bhh,
                        float*       __restrict__ hbuf,   // [2][HIDDEN]
                        unsigned*    __restrict__ bar,
                        float*       __restrict__ out) {  // [3*HIDDEN]
    __shared__ float h_s[HIDDEN];
    __shared__ float gpart[4 * 128];
    __shared__ float gfin[128];

    const int b    = blockIdx.x;
    const int tid  = threadIdx.x;
    const int wave = tid >> 5;
    const int lane = tid & 31;
    const int p    = wave & 3;          // k-part: k in [128p, 128p+128)
    const int gate = wave >> 2;         // 0..3 = i,f,g,o
    const int lr   = (gate << 5) + lane;            // local row 0..127
    const int grow = (gate << 9) + (b << 5) + lane; // global gate row

    // --- W_hh rows -> registers (one-time) ---
    float4 wreg[32];
    {
        const float4* wsrc = (const float4*)(Whh + (size_t)grow * HIDDEN + (p << 7));
#pragma unroll
        for (int i = 0; i < 32; ++i) wreg[i] = wsrc[i];
    }

    // --- per-thread constants for the reduction stage (threads 0..127) ---
    float bh2 = 0.0f; int gr = 0;
    if (tid < 128) {
        gr  = ((tid >> 5) << 9) + (b << 5) + (tid & 31);
        bh2 = bhh[gr];
    }

    // --- LSTM state owned by threads 0..31 (j = 32b + tid) ---
    const int j = (b << 5) + tid;
    float c_st = 0.0f, h_st = 0.0f;
    if (tid < 32) { c_st = c0[j]; h_st = h0[j]; }

    // LDS byte offset of this thread's h_s slot (gfx1250: low 32 bits of a
    // generic LDS address are the LDS offset).
    const unsigned lds_off = (unsigned)(size_t)&h_s[tid];

    for (int t = 0; t < SEQ; ++t) {
        // async broadcast of previous h into LDS (double buffer by t&1)
        {
            unsigned voff = (unsigned)((((t & 1) << 9) + tid) * 4);
            asm volatile(
                "global_load_async_to_lds_b32 %0, %1, %2\n\t"
                "s_wait_asynccnt 0"
                :: "v"(lds_off), "v"(voff), "s"(hbuf)
                : "memory");
        }
        __syncthreads();

        // prefetch next step's x-gate row while we compute
        if (tid < 128 && (t + 1) < SEQ)
            __builtin_prefetch(&xg[(size_t)(t + 1) * GATES + gr], 0, 1);

        // 128-wide partial dot: uniform-address float4 LDS reads (broadcast)
        const float4* h4 = (const float4*)h_s + (p << 5);
        float s = 0.0f;
#pragma unroll
        for (int i = 0; i < 32; ++i) {
            float4 hv = h4[i];
            s = fmaf(wreg[i].x, hv.x, s);
            s = fmaf(wreg[i].y, hv.y, s);
            s = fmaf(wreg[i].z, hv.z, s);
            s = fmaf(wreg[i].w, hv.w, s);
        }
        gpart[(p << 7) + lr] = s;
        __syncthreads();

        // reduce 4 k-parts + b_hh + precomputed x-gate
        if (tid < 128)
            gfin[tid] = gpart[tid] + gpart[128 + tid] + gpart[256 + tid] +
                        gpart[384 + tid] + bh2 + xg[(size_t)t * GATES + gr];
        __syncthreads();

        // gate math + state update on the 32 state threads
        if (tid < 32) {
            const bool pad = (tok[t] == PAD_IDX);
            float gi = gfin[tid];
            float gf = gfin[32 + tid];
            float gg = gfin[64 + tid];
            float go = gfin[96 + tid];
            float cn = sigm(gf) * c_st + sigm(gi) * tanhf(gg);
            float hn = sigm(go) * tanhf(cn);
            if (!pad) { c_st = cn; h_st = hn; }   // PAD leaves state untouched
            hbuf[(((t + 1) & 1) << 9) + j] = h_st;
        }
        __threadfence();
        __syncthreads();

        // grid-wide step barrier: monotonic counter, release/acquire
        if (tid == 0) {
            __hip_atomic_fetch_add(bar, 1u, __ATOMIC_RELEASE, __HIP_MEMORY_SCOPE_AGENT);
            const unsigned target = (unsigned)(t + 1) * (unsigned)NBLK;
            while (__hip_atomic_load(bar, __ATOMIC_ACQUIRE, __HIP_MEMORY_SCOPE_AGENT) < target)
                __builtin_amdgcn_s_sleep(2);
        }
        __syncthreads();
        __threadfence();
    }

    // d_out = [out, h, c]; with h0 == 0 the reference's out equals final h.
    if (tid < 32) {
        out[j]              = h_st;
        out[HIDDEN + j]     = h_st;
        out[2 * HIDDEN + j] = c_st;
    }
}

// ============================================================================
// Host launcher
// ============================================================================
extern "C" void kernel_launch(void* const* d_in, const int* in_sizes, int n_in,
                              void* d_out, int out_size, void* d_ws, size_t ws_size,
                              hipStream_t stream) {
    const int*   tok = (const int*)  d_in[0];   // input_seq [1,1024] int32
    const float* h0  = (const float*)d_in[1];   // [1,1,512]
    const float* c0  = (const float*)d_in[2];   // [1,1,512]
    const float* emb = (const float*)d_in[3];   // [100000,300]
    const float* Wih = (const float*)d_in[4];   // [2048,300]
    const float* Whh = (const float*)d_in[5];   // [2048,512]
    const float* bih = (const float*)d_in[6];   // [2048]
    const float* bhh = (const float*)d_in[7];   // [2048]
    float* out = (float*)d_out;                 // [3*512]

    // workspace layout
    char* ws = (char*)d_ws;
    size_t o = 0;
    _Float16* Wf = (_Float16*)(ws + o); o += (size_t)KPAD * GATES * sizeof(_Float16); // 1.25 MB
    _Float16* xs = (_Float16*)(ws + o); o += (size_t)SEQ  * KPAD  * sizeof(_Float16); // 0.63 MB
    float*    xg = (float*)   (ws + o); o += (size_t)SEQ  * GATES * sizeof(float);    // 8 MB
    float*  hbuf = (float*)   (ws + o); o += (size_t)2 * HIDDEN * sizeof(float);
    unsigned* bar = (unsigned*)(ws + o);

    k_pack_wih<<<(KPAD * GATES + 255) / 256, 256, 0, stream>>>(Wih, Wf);
    k_gather  <<<(SEQ * KPAD + 255) / 256, 256, 0, stream>>>(tok, emb, xs);
    k_xgates  <<<(SEQ / 16) * (GATES / 16) / 4 / 8, 256, 0, stream>>>(xs, Wf, bih, xg);
    k_init    <<<1, 512, 0, stream>>>(h0, hbuf, bar);
    k_recur   <<<NBLK, RTHREADS, 0, stream>>>(tok, h0, c0, xg, Whh, bhh, hbuf, bar, out);
}
